// VectorQuantizer_73839077753453
// MI455X (gfx1250) — compile-verified, matrix operand order checked by reading, not run on hardware
//
#include <hip/hip_runtime.h>

// VQ-VAE vector quantizer for MI455X (gfx1250), round 4.
// argmin_n ||x - e_n||^2 via score = ||e||^2 - 2 x.e ; gather ; loss = 1.25*MSE.
// - GEMM: split-bf16 (x = hi+lo, 2-term Ozaki), 3x V_WMMA_F32_16X16X32_BF16 per
//   32-wide k-chunk, independent accumulator chains, f32 accumulate.
// - Staging: GLOBAL_LOAD_ASYNC_TO_LDS_B128 (ASYNCcnt) into double-buffered,
//   bank-conflict-free LDS; 2 tiles (8 KB) per barrier iteration to halve
//   workgroup sync overhead. s_wait_asynccnt 0 before each barrier.
// - E pre-split once into bf16 hi/lo planes + ||e||^2 (prep kernel).
// - Loss recomputed exactly in f32 during the gather epilogue.

#define K_EMB   8192
#define DIM     64
#define N_VEC   32768
#define TILES   (K_EMB / 16)          // 512 embedding tiles of 16
#define TPI     2                     // tiles per barrier iteration
#define ITERS   (TILES / TPI)         // 256
#define WAVES   8
#define ROWS_WG (WAVES * 16)          // 128 rows per workgroup
#define NWG     (N_VEC / ROWS_WG)     // 256 workgroups
#define BROW    72                    // ushorts per LDS plane row (144 B = 36 dwords)
#define PLANE   (16 * BROW)           // 1152 ushorts per 16x64 bf16 plane
#define EPLANE_BYTES ((unsigned)(K_EMB * DIM * 2))   // 1 MB per bf16 plane

typedef float          v2f    __attribute__((ext_vector_type(2)));
typedef float          v8f    __attribute__((ext_vector_type(8)));
typedef __bf16         v16bf  __attribute__((ext_vector_type(16)));
typedef unsigned short u16x8  __attribute__((ext_vector_type(8)));
typedef unsigned short u16x16 __attribute__((ext_vector_type(16)));

static __device__ __forceinline__ unsigned short bf16_rne(float v) {
    unsigned u = __float_as_uint(v);
    return (unsigned short)((u + 0x7FFFu + ((u >> 16) & 1u)) >> 16);
}

// ---- Prep: split E into bf16 hi/lo planes and compute ||e||^2 (once) ----
__global__ __launch_bounds__(64, 1)
void vq_prep(const float* __restrict__ emb, unsigned short* __restrict__ ehi,
             unsigned short* __restrict__ elo, float* __restrict__ esq)
{
    __shared__ float sm[2];
    const int r = blockIdx.x;
    const int k = threadIdx.x;                 // 0..63
    const float v = emb[(size_t)r * DIM + k];
    const unsigned short h = bf16_rne(v);
    const float hf = __uint_as_float(((unsigned)h) << 16);
    const unsigned short l = bf16_rne(v - hf);
    ehi[(size_t)r * DIM + k] = h;
    elo[(size_t)r * DIM + k] = l;
    float p = v * v;
#pragma unroll
    for (int off = 16; off > 0; off >>= 1) p += __shfl_down(p, off, 32);
    if ((k & 31) == 0) sm[k >> 5] = p;
    __syncthreads();
    if (k == 0) esq[r] = sm[0] + sm[1];
}

// ---- Main: distances + argmin + gather + per-wave loss partial ----
__global__ __launch_bounds__(256, 1)
void vq_argmin_kernel(const float* __restrict__ x,
                      const unsigned short* __restrict__ ehi,   // hi plane; lo at +1MB
                      const float* __restrict__ esq,
                      const float* __restrict__ emb,
                      float* __restrict__ qout,
                      float* __restrict__ idxout,
                      float* __restrict__ wloss)
{
    __shared__ unsigned short ldsb[2 * TPI * 2 * PLANE]; // [buf][tt][plane][16][72]
    __shared__ float ldse[2 * TPI * 16];                 // staged ||e||^2

    const int tid   = threadIdx.x;
    const int wave  = tid >> 5;
    const int lane  = tid & 31;
    const int nloc  = lane & 15;
    const int khalf = lane >> 4;
    const int rowbase = blockIdx.x * ROWS_WG + wave * 16;
    const int myrow   = rowbase + nloc;

    const unsigned long long ehi_base = (unsigned long long)ehi;  // uniform -> SGPR pair
    const unsigned long long esq_base = (unsigned long long)esq;

    // Resident A fragments (ISA 16-bit A 16x32 layout): lane<16 holds row nloc,
    // K in [0,8)U[16,24); lane>=16 holds K in [8,16)U[24,32); chunk c adds 32c.
    v16bf a_hi[2], a_lo[2];
    {
        const float* xr = x + (size_t)myrow * DIM;
#pragma unroll
        for (int c = 0; c < 2; ++c) {
            u16x16 hb, lb;
#pragma unroll
            for (int i = 0; i < 16; ++i) {
                const int k = 32 * c + 8 * khalf + i + ((i >= 8) ? 8 : 0);
                const float v = xr[k];
                const unsigned short h = bf16_rne(v);
                const float hf = __uint_as_float(((unsigned)h) << 16);
                const unsigned short l = bf16_rne(v - hf);
                hb[i] = h; lb[i] = l;
            }
            a_hi[c] = __builtin_bit_cast(v16bf, hb);
            a_lo[c] = __builtin_bit_cast(v16bf, lb);
        }
    }

    float best_s[8];
    int   best_i[8];
#pragma unroll
    for (int v = 0; v < 8; ++v) { best_s[v] = __builtin_inff(); best_i[v] = 0; }

    // Stage TPI tiles (8 KB) via async global->LDS DMA (no VGPR round-trip).
    // Thread layout: plane = tid>>7 (hi/lo), r = (tid>>3)&15, c = (tid&7)*8 ushorts.
    auto stage = [&](int pair, int buf) {
        const int plane = tid >> 7;
        const int idx   = tid & 127;
        const int r     = idx >> 3;
        const int c     = (idx & 7) << 3;
#pragma unroll
        for (int tt = 0; tt < TPI; ++tt) {
            const unsigned goff = (unsigned)plane * EPLANE_BYTES +
                ((unsigned)((pair * TPI + tt) * 16 + r) * DIM + c) * 2u;
            const unsigned loff = (unsigned)(size_t)
                &ldsb[((buf * TPI + tt) * 2 + plane) * PLANE + r * BROW + c];
            asm volatile("global_load_async_to_lds_b128 %0, %1, %2"
                         :: "v"(loff), "v"(goff), "s"(ehi_base) : "memory");
        }
        if (tid < 32) {   // wave 0 only: uniform branch, full EXEC
            const unsigned ge = (unsigned)(pair * (TPI * 16) + tid) * 4u;
            const unsigned le = (unsigned)(size_t)&ldse[buf * (TPI * 16) + tid];
            asm volatile("global_load_async_to_lds_b32 %0, %1, %2"
                         :: "v"(le), "v"(ge), "s"(esq_base) : "memory");
        }
    };

    stage(0, 0);

    for (int i = 0; i < ITERS; ++i) {
        const int buf = i & 1;
        asm volatile("s_wait_asynccnt 0x0" ::: "memory");  // staged data landed in LDS
        __syncthreads();                                   // ... and visible to all waves
        if (i + 1 < ITERS) stage(i + 1, buf ^ 1);

#pragma unroll
        for (int tt = 0; tt < TPI; ++tt) {
            // B fragments from LDS (32x16 bf16 layout == same gather as A, N = nloc)
            const unsigned short* bh =
                &ldsb[((buf * TPI + tt) * 2 + 0) * PLANE + nloc * BROW + 8 * khalf];
            const unsigned short* bl =
                &ldsb[((buf * TPI + tt) * 2 + 1) * PLANE + nloc * BROW + 8 * khalf];
            const float esqv = ldse[buf * (TPI * 16) + tt * 16 + nloc];

            // Two independent 3-WMMA accumulation chains (one per 32-wide k-chunk)
            v8f acc0 = {}, acc1 = {};
#pragma unroll
            for (int c = 0; c < 2; ++c) {
                u16x8 h0 = *(const u16x8*)(bh + 32 * c);
                u16x8 h1 = *(const u16x8*)(bh + 32 * c + 16);
                u16x8 l0 = *(const u16x8*)(bl + 32 * c);
                u16x8 l1 = *(const u16x8*)(bl + 32 * c + 16);
                v16bf bhi = __builtin_bit_cast(v16bf, __builtin_shufflevector(
                    h0, h1, 0,1,2,3,4,5,6,7,8,9,10,11,12,13,14,15));
                v16bf blo = __builtin_bit_cast(v16bf, __builtin_shufflevector(
                    l0, l1, 0,1,2,3,4,5,6,7,8,9,10,11,12,13,14,15));
                v8f& acc = c ? acc1 : acc0;
                // (ahi+alo)(bhi+blo) ~= ahi*bhi + ahi*blo + alo*bhi (lo*lo ~2^-16 rel)
                acc = __builtin_amdgcn_wmma_f32_16x16x32_bf16(
                    false, a_hi[c], false, bhi, (short)0, acc, false, false);
                acc = __builtin_amdgcn_wmma_f32_16x16x32_bf16(
                    false, a_hi[c], false, blo, (short)0, acc, false, false);
                acc = __builtin_amdgcn_wmma_f32_16x16x32_bf16(
                    false, a_lo[c], false, bhi, (short)0, acc, false, false);
            }

            const int gidx = (i * TPI + tt) * 16 + nloc;
#pragma unroll
            for (int v = 0; v < 8; ++v) {
                // score = ||e||^2 - 2*(dot_chunk0 + dot_chunk1)
                const float s = fmaf(-2.0f, acc0[v], fmaf(-2.0f, acc1[v], esqv));
                if (s < best_s[v]) { best_s[v] = s; best_i[v] = gidx; }
            }
        }
    }

    // Argmin butterfly across the 16 N-lanes of each C-matrix half
#pragma unroll
    for (int off = 1; off < 16; off <<= 1) {
#pragma unroll
        for (int v = 0; v < 8; ++v) {
            const float os = __shfl_xor(best_s[v], off, 32);
            const int   oi = __shfl_xor(best_i[v], off, 32);
            if (os < best_s[v] || (os == best_s[v] && oi < best_i[v])) {
                best_s[v] = os; best_i[v] = oi;
            }
        }
    }
    // Row m = v + 8*khalf: result replicated in lanes 0..15 (m<8) / 16..31 (m>=8)

    // Gather quantized rows + indices + exact-f32 loss partial
    float wl = 0.0f;
#pragma unroll
    for (int m = 0; m < 16; ++m) {
        const int v   = m & 7;
        const int src = (m < 8) ? 0 : 16;
        const int im  = __shfl(best_i[v], src, 32);
        v2f e2 = *(const v2f*)(emb + (size_t)im * DIM + 2 * lane);
        v2f xv = *(const v2f*)(x + (size_t)(rowbase + m) * DIM + 2 * lane);
        const float dx = e2.x - xv.x, dy = e2.y - xv.y;
        wl = fmaf(dx, dx, fmaf(dy, dy, wl));
        *(v2f*)(qout + (size_t)(rowbase + m) * DIM + 2 * lane) = e2;
        if (lane == 0) idxout[rowbase + m] = (float)im;
    }
#pragma unroll
    for (int off = 16; off > 0; off >>= 1) wl += __shfl_down(wl, off, 32);
    if (lane == 0) wloss[blockIdx.x * WAVES + wave] = wl;
}

__global__ __launch_bounds__(256, 1)
void vq_loss_reduce(const float* __restrict__ wloss, float* __restrict__ out_loss)
{
    __shared__ float sm[8];
    const int tid = threadIdx.x;
    float s = 0.0f;
    for (int i = tid; i < NWG * WAVES; i += 256) s += wloss[i];
#pragma unroll
    for (int off = 16; off > 0; off >>= 1) s += __shfl_down(s, off, 32);
    if ((tid & 31) == 0) sm[tid >> 5] = s;
    __syncthreads();
    if (tid == 0) {
        float t = 0.0f;
#pragma unroll
        for (int i = 0; i < 8; ++i) t += sm[i];
        // loss = (1 + 0.25) * mean((q-x)^2) over 32*1024*64 elements
        out_loss[0] = 1.25f * t / 2097152.0f;
    }
}

extern "C" void kernel_launch(void* const* d_in, const int* in_sizes, int n_in,
                              void* d_out, int out_size, void* d_ws, size_t ws_size,
                              hipStream_t stream)
{
    const float* x   = (const float*)d_in[0];   // [32,1024,64] f32
    const float* emb = (const float*)d_in[1];   // [8192,64]   f32

    float* out    = (float*)d_out;
    float* qout   = out;                        // 2097152 floats
    float* loss   = out + (size_t)N_VEC * DIM;  // 1 float
    float* idxout = loss + 1;                   // 32768 indices (as float values)

    // Workspace: esq(32KB) | wloss(8KB) | E_hi(1MB bf16) | E_lo(1MB bf16, contiguous)
    float*          esqw   = (float*)d_ws;
    float*          wlossw = esqw + K_EMB;
    unsigned short* ehi    = (unsigned short*)(wlossw + NWG * WAVES);
    unsigned short* elo    = ehi + (size_t)K_EMB * DIM;   // == ehi + 1MB (contiguous)

    vq_prep<<<K_EMB, 64, 0, stream>>>(emb, ehi, elo, esqw);
    vq_argmin_kernel<<<NWG, 256, 0, stream>>>(x, ehi, esqw, emb,
                                              qout, idxout, wlossw);
    vq_loss_reduce<<<1, 256, 0, stream>>>(wlossw, loss);
}
